// liBRU_28664611733942
// MI455X (gfx1250) — matrix operand order repaired and checked by reading, not compiled, not run on hardware
//
#include <hip/hip_runtime.h>
#include <hip/hip_bf16.h>

// ---------------------------------------------------------------------------
// LiBRU on MI455X (gfx1250, wave32, WMMA + async-to-LDS).
//   B=32, T=1000, F=256, H=512.
// Phase 0: cast W,Wz,V,Vz fp32 -> bf16 (workspace).
// Phase 1: Wx = LN(x W^T), Wzx = LN(x Wz^T) via v_wmma_f32_16x16x32_bf16;
//          x tile staged with GLOBAL_LOAD_ASYNC_TO_LDS_B128; fused LayerNorm
//          (one wave32 per row); outputs stored bf16 in workspace.
// Phase 2: recurrent scan. Batch rows independent -> 2 persistent workgroups
//          (M=16 each). h state: bf16 WMMA operand in LDS + fp32 log-space
//          copy in registers. Per-step: async-stage Wx_t/Wzx_t tiles into LDS
//          (overlapping the GEMM), two h·V^T GEMMs with WMMA bf16,
//          s_wait_asynccnt + barrier, elementwise BRU update.
// ---------------------------------------------------------------------------

#define LB_B 32
#define LB_T 1000
#define LB_F 256
#define LB_H 512

typedef __attribute__((ext_vector_type(16))) __bf16 v16bf;
typedef __attribute__((ext_vector_type(8)))  float  v8f;

union PackBF2 { unsigned int u; __bf16 h[2]; };

static __device__ inline v8f wmma_bf16(v16bf a, v16bf b, v8f c) {
    // D = A(16x32) * B(32x16) + C(16x16 f32)
    return __builtin_amdgcn_wmma_f32_16x16x32_bf16(
        /*neg_a=*/false, a, /*neg_b=*/false, b,
        /*c_mod=*/(short)0, c, /*reuse_a=*/false, /*reuse_b=*/false);
}

// 16-byte async copy global -> LDS (ASYNCcnt path). GV addressing mode.
static __device__ inline void async_copy16(unsigned lds_addr, const void* gptr) {
    asm volatile("global_load_async_to_lds_b128 %0, %1, off"
                 :: "v"(lds_addr), "v"((unsigned long long)(size_t)gptr)
                 : "memory");
}
static __device__ inline void wait_async0() {
    asm volatile("s_wait_asynccnt 0x0" ::: "memory");
}

// A-fragment (16x32 bf16, MxK) gathered from an LDS tile of row stride KDIM.
// Contiguous k order used here: lanes 0-15 -> M=lane, K=0..15; lanes 16-31 -> same
// M, K=16..31. Pairs contiguous -> compiler fuses to ds_load_b128.
template <int KDIM>
static __device__ inline v16bf load_a_frag(const __bf16 (*s)[KDIM], int kk, int lane) {
    const int m    = lane & 15;
    const int half = lane >> 4;
    v16bf a;
#pragma unroll
    for (int p = 0; p < 8; ++p) {
        const int k = kk * 32 + half * 16 + p * 2;
        PackBF2 pk;
        pk.u = *(const unsigned int*)(&s[m][k]);
        a[2 * p]     = pk.h[0];
        a[2 * p + 1] = pk.h[1];
    }
    return a;
}

// B-fragment (32x16 bf16, KxN) for out = h * M^T with M stored row-major
// [N][K]: column n of B == row n of M, contiguous in K.
// Contiguous k order (same as A): lanes 0-15 -> N=lane, K=0..15; lanes 16-31 -> N=lane-16, K=16..31.
static __device__ inline v16bf load_b_frag(const __bf16* __restrict__ base, int ldk,
                                           int ntile, int kk, int lane) {
    const int nl = lane & 15;
    const int k0 = kk * 32 + (lane >> 4) * 16;
    return *(const v16bf*)(base + (size_t)(ntile * 16 + nl) * ldk + k0);
}

// ---------------------------------------------------------------------------
// Phase 0: weight casts fp32 -> bf16
// ---------------------------------------------------------------------------
__global__ void cast_weights_kernel(const float* __restrict__ W,  const float* __restrict__ Wz,
                                    const float* __restrict__ V,  const float* __restrict__ Vz,
                                    __bf16* __restrict__ Wbf,  __bf16* __restrict__ Wzbf,
                                    __bf16* __restrict__ Vbf,  __bf16* __restrict__ Vzbf) {
    const int i = blockIdx.x * blockDim.x + threadIdx.x;
    const int NW = LB_H * LB_F;   // 131072
    const int NV = LB_H * LB_H;   // 262144
    if (i < NW) { Wbf[i] = (__bf16)W[i];  Wzbf[i] = (__bf16)Wz[i]; }
    if (i < NV) { Vbf[i] = (__bf16)V[i];  Vzbf[i] = (__bf16)Vz[i]; }
}

// ---------------------------------------------------------------------------
// Phase 1: projections + fused LayerNorm.
// grid = (B*T)/16 = 2000 blocks, 512 threads (16 waves).
// ---------------------------------------------------------------------------
__global__ __launch_bounds__(512)
void proj_ln_kernel(const float* __restrict__ x,
                    const __bf16* __restrict__ Wbf, const __bf16* __restrict__ Wzbf,
                    const float* __restrict__ gamma, const float* __restrict__ beta,
                    __bf16* __restrict__ Wxbf, __bf16* __restrict__ Wzxbf) {
    __shared__ __align__(16) float  sxf[16][LB_F];   // 16 KB raw x tile (async staged)
    __shared__ __align__(16) __bf16 sx [16][LB_F];   //  8 KB bf16 A operand
    __shared__ __align__(16) float  sout[16][LB_H];  // 32 KB pre-LN GEMM tile

    const int tid  = threadIdx.x;
    const int lane = tid & 31;
    const int wave = tid >> 5;
    const int row0 = blockIdx.x * 16;   // flattened row index b*T + t

    // Async-stage the contiguous 16KB x tile into LDS (fp32).
    {
        const unsigned sxf_base = (unsigned)(size_t)&sxf[0][0];
        const char* xB = (const char*)x + (size_t)row0 * LB_F * 4;
#pragma unroll
        for (int j = 0; j < 2; ++j) {
            const int e = tid + 512 * j;               // 1024 x 16B transfers
            async_copy16(sxf_base + e * 16, xB + (size_t)e * 16);
        }
    }
    wait_async0();
    __syncthreads();

    // Convert to bf16 in LDS.
#pragma unroll
    for (int j = 0; j < 8; ++j) {
        const int e = tid + 512 * j;                   // 16*256 = 4096 elements
        const int r = e >> 8, c = e & 255;
        sx[r][c] = (__bf16)sxf[r][c];
    }
    __syncthreads();

    const int nt0 = 2 * wave, nt1 = 2 * wave + 1;      // 32 N-tiles over 16 waves
    const int mbase = (lane >> 4) * 8;
    const int nl = lane & 15;

    for (int mat = 0; mat < 2; ++mat) {
        const __bf16* Wm  = mat ? Wzbf  : Wbf;
        __bf16*       dst = mat ? Wzxbf : Wxbf;

        v8f acc0 = {}, acc1 = {};
#pragma unroll
        for (int kk = 0; kk < LB_F / 32; ++kk) {
            const v16bf a  = load_a_frag<LB_F>(sx, kk, lane);
            const v16bf b0 = load_b_frag(Wm, LB_F, nt0, kk, lane);
            const v16bf b1 = load_b_frag(Wm, LB_F, nt1, kk, lane);
            acc0 = wmma_bf16(a, b0, acc0);
            acc1 = wmma_bf16(a, b1, acc1);
        }

        // Scatter C fragments (lane->N, reg r->M, +8 for upper lane half).
#pragma unroll
        for (int r = 0; r < 8; ++r) {
            sout[mbase + r][nt0 * 16 + nl] = acc0[r];
            sout[mbase + r][nt1 * 16 + nl] = acc1[r];
        }
        __syncthreads();

        // LayerNorm over H=512: wave `wave` owns row `wave`.
        float s = 0.f, ss = 0.f;
#pragma unroll
        for (int j = 0; j < 16; ++j) {
            const float v = sout[wave][lane + 32 * j];
            s += v; ss += v * v;
        }
#pragma unroll
        for (int off = 16; off > 0; off >>= 1) {
            s  += __shfl_xor(s,  off, 32);
            ss += __shfl_xor(ss, off, 32);
        }
        const float mu   = s * (1.f / (float)LB_H);
        const float var  = ss * (1.f / (float)LB_H) - mu * mu;
        const float rstd = rsqrtf(var + 1e-5f);
#pragma unroll
        for (int j = 0; j < 16; ++j) {
            const int c = lane + 32 * j;
            const float v = (sout[wave][c] - mu) * rstd * gamma[c] + beta[c];
            dst[(size_t)(row0 + wave) * LB_H + c] = (__bf16)v;
        }
        __syncthreads();   // protect sout before next matrix reuses it
    }
}

// ---------------------------------------------------------------------------
// Phase 2: sequential scan. Batch rows independent -> grid = 2 blocks of
// 16 batch rows, 512 threads (16 waves). h bf16 in LDS, h fp32 in registers.
// Wx_t / Wzx_t tiles async-staged into LDS each step, overlapping the GEMM.
// ---------------------------------------------------------------------------
__global__ __launch_bounds__(512)
void scan_kernel(const __bf16* __restrict__ Vbf,  const __bf16* __restrict__ Vzbf,
                 const __bf16* __restrict__ Wxbf, const __bf16* __restrict__ Wzxbf,
                 float* __restrict__ out) {
    __shared__ __align__(16) __bf16 sh  [16][LB_H];  // 16 KB: h as WMMA A operand
    __shared__ __align__(16) __bf16 swx [16][LB_H];  // 16 KB: Wx_t tile (async)
    __shared__ __align__(16) __bf16 swzx[16][LB_H];  // 16 KB: Wzx_t tile (async)

    const int tid  = threadIdx.x;
    const int lane = tid & 31;
    const int wave = tid >> 5;
    const int b0   = blockIdx.x * 16;   // batch rows owned by this workgroup

    // h0 = 0 (bf16 operand in LDS).
#pragma unroll
    for (int j = 0; j < 16; ++j) {
        const int e = tid + 512 * j;    // 16*512 = 8192 elements
        sh[e >> 9][e & 511] = (__bf16)0.f;
    }
    __syncthreads();

    const int nt0 = 2 * wave, nt1 = 2 * wave + 1;
    const int mbase = (lane >> 4) * 8;
    const int nl = lane & 15;

    const unsigned swx_base  = (unsigned)(size_t)&swx[0][0];
    const unsigned swzx_base = (unsigned)(size_t)&swzx[0][0];
    const char* wxB  = (const char*)Wxbf;
    const char* wzxB = (const char*)Wzxbf;

    float hreg[2][8];                   // fp32 log-space h owned by this lane
#pragma unroll
    for (int ti = 0; ti < 2; ++ti)
#pragma unroll
        for (int r = 0; r < 8; ++r) hreg[ti][r] = 0.f;

    for (int t = 0; t < LB_T; ++t) {
        // ----- async-stage Wx_t / Wzx_t tiles (16 rows x 1KB each) into LDS.
        // Issued before the GEMM so the copy overlaps compute.
#pragma unroll
        for (int j = 0; j < 2; ++j) {
            const int e  = tid + 512 * j;            // 1024 x 16B per matrix
            const int mm = e >> 6, c16 = e & 63;
            const size_t gb = ((size_t)(b0 + mm) * LB_T + t) * (LB_H * 2)
                              + (size_t)c16 * 16;
            async_copy16(swx_base  + e * 16, wxB  + gb);
            async_copy16(swzx_base + e * 16, wzxB + gb);
        }

        // Warm L2 for the next step's tiles (global_prefetch_b8).
        if (t + 1 < LB_T) {
            const size_t pre = ((size_t)(b0 + mbase) * LB_T + (t + 1)) * LB_H
                               + nt0 * 16 + nl;
            __builtin_prefetch(&Wxbf[pre],  0, 1);
            __builtin_prefetch(&Wzxbf[pre], 0, 1);
        }

        // ----- GEMM phase: accz = h Vz^T, accc = h V^T (this wave's 2 N-tiles)
        v8f accz[2] = { v8f{}, v8f{} };
        v8f accc[2] = { v8f{}, v8f{} };
        for (int kk = 0; kk < LB_H / 32; ++kk) {
            const v16bf a   = load_a_frag<LB_H>(sh, kk, lane);
            const v16bf bz0 = load_b_frag(Vzbf, LB_H, nt0, kk, lane);
            const v16bf bc0 = load_b_frag(Vbf,  LB_H, nt0, kk, lane);
            const v16bf bz1 = load_b_frag(Vzbf, LB_H, nt1, kk, lane);
            const v16bf bc1 = load_b_frag(Vbf,  LB_H, nt1, kk, lane);
            accz[0] = wmma_bf16(a, bz0, accz[0]);
            accc[0] = wmma_bf16(a, bc0, accc[0]);
            accz[1] = wmma_bf16(a, bz1, accz[1]);
            accc[1] = wmma_bf16(a, bc1, accc[1]);
        }

        wait_async0();       // staged Wx/Wzx tiles landed (this wave's copies)
        __syncthreads();     // all waves: GEMM reads of sh done + tiles visible

        // ----- elementwise log-space BRU update (each (m,n) owned by 1 lane)
#pragma unroll
        for (int ti = 0; ti < 2; ++ti) {
            const int ntile = 2 * wave + ti;
            const int n = ntile * 16 + nl;
#pragma unroll
            for (int r = 0; r < 8; ++r) {
                const int mm = mbase + r;                        // local row
                const float wz = (float)swzx[mm][n];
                const float wx = (float)swx [mm][n];
                const float az = accz[ti][r] + wz;
                const float z  = 1.f / (1.f + expf(-az));
                const float cp = accc[ti][r] + wx;
                const float ct = (cp > 20.f) ? cp : log1pf(expf(cp));
                const float hold = hreg[ti][r];
                const float hnew = logf(z * expf(hold) + (1.f - z) * expf(ct));
                hreg[ti][r] = hnew;
                out[((size_t)(b0 + mm) * LB_T + t) * LB_H + n] = hnew;
                sh[mm][n] = (__bf16)hnew;
            }
        }
        __syncthreads();     // h(t+1) visible before next GEMM; staging buffers
                             // free for the next step's async copies
    }
}

// ---------------------------------------------------------------------------
// Launch
// ---------------------------------------------------------------------------
extern "C" void kernel_launch(void* const* d_in, const int* in_sizes, int n_in,
                              void* d_out, int out_size, void* d_ws, size_t ws_size,
                              hipStream_t stream) {
    (void)in_sizes; (void)n_in; (void)out_size; (void)ws_size;

    const float* x     = (const float*)d_in[0];
    const float* W     = (const float*)d_in[1];
    const float* Wz    = (const float*)d_in[2];
    const float* V     = (const float*)d_in[3];
    const float* Vz    = (const float*)d_in[4];
    const float* gamma = (const float*)d_in[5];
    const float* beta  = (const float*)d_in[6];
    float* out = (float*)d_out;

    // Workspace layout (all offsets 256B-aligned by construction).
    char* ws = (char*)d_ws;
    size_t off = 0;
    __bf16* Wbf   = (__bf16*)(ws + off); off += (size_t)LB_H * LB_F * 2;          // 256 KB
    __bf16* Wzbf  = (__bf16*)(ws + off); off += (size_t)LB_H * LB_F * 2;          // 256 KB
    __bf16* Vbf   = (__bf16*)(ws + off); off += (size_t)LB_H * LB_H * 2;          // 512 KB
    __bf16* Vzbf  = (__bf16*)(ws + off); off += (size_t)LB_H * LB_H * 2;          // 512 KB
    __bf16* Wxbf  = (__bf16*)(ws + off); off += (size_t)LB_B * LB_T * LB_H * 2;   // 32.77 MB
    __bf16* Wzxbf = (__bf16*)(ws + off); off += (size_t)LB_B * LB_T * LB_H * 2;   // 32.77 MB

    // Phase 0: weight casts (262144 elements max -> 1024 blocks x 256).
    cast_weights_kernel<<<1024, 256, 0, stream>>>(W, Wz, V, Vz, Wbf, Wzbf, Vbf, Vzbf);

    // Phase 1: projections + LN. 32000 rows / 16 = 2000 blocks.
    proj_ln_kernel<<<(LB_B * LB_T) / 16, 512, 0, stream>>>(x, Wbf, Wzbf, gamma, beta,
                                                           Wxbf, Wzxbf);

    // Phase 2: scan. 2 persistent workgroups (one per 16 batch rows).
    scan_kernel<<<LB_B / 16, 512, 0, stream>>>(Vbf, Vzbf, Wxbf, Wzxbf, out);
}
